// GALA_42125039239630
// MI455X (gfx1250) — compile-verified, matrix operand order checked by reading, not run on hardware
//
#include <hip/hip_runtime.h>
#include <hip/hip_bf16.h>

// ---------------------------------------------------------------------------
// GCN autoencoder: 4 x (GEMM + SpMM scatter-add + bias + ReLU)
// GEMMs via v_wmma_f32_16x16x32_bf16 (CDNA5), all shapes compile-time.
// SpMM via float4 gathers + L2 float atomics.
// ---------------------------------------------------------------------------

typedef __attribute__((ext_vector_type(16))) __bf16 v16bf;
typedef __attribute__((ext_vector_type(8)))  float  v8f;

// ---------------- elementwise helpers ----------------

__global__ void k_fill4_f32(float4* __restrict__ p, float v, long long n4) {
  long long i = (long long)blockIdx.x * blockDim.x + threadIdx.x;
  if (i < n4) p[i] = make_float4(v, v, v, v);
}

__global__ void k_count_deg(const int* __restrict__ dst, float* __restrict__ deg, int E) {
  int e = blockIdx.x * blockDim.x + threadIdx.x;
  if (e < E) atomicAdd(&deg[dst[e]], 1.0f);
}

__global__ void k_norms(const float* __restrict__ deg,
                        float* __restrict__ dis, float* __restrict__ dls,
                        float* __restrict__ aself, float* __restrict__ lself, int N) {
  int v = blockIdx.x * blockDim.x + threadIdx.x;
  if (v >= N) return;
  float d  = deg[v];                 // >= 1 (self-loop counted at init)
  float di = (d > 0.0f) ? rsqrtf(d) : 0.0f;
  float dl = rsqrtf(d + 1.0f);
  dis[v] = di;
  dls[v] = dl;
  aself[v] = di * di;                // GCN self-loop weight
  lself[v] = 2.0f * dl * dl;         // Laplacian self-loop weight (+2 diagonal)
}

__global__ void k_edge_vals(const int* __restrict__ src, const int* __restrict__ dst,
                            const float* __restrict__ dis, const float* __restrict__ dls,
                            float* __restrict__ adjv, float* __restrict__ lapv, int E) {
  int e = blockIdx.x * blockDim.x + threadIdx.x;
  if (e >= E) return;
  int s = src[e], d = dst[e];
  adjv[e] = dis[s] * dis[d];
  float w = (s == d) ? 2.0f : -1.0f;   // accidental self-loops get +2
  lapv[e] = dls[s] * w * dls[d];
}

// ---------------- dense GEMM: H = X @ W  (bf16 WMMA, f32 accumulate) ----------------
// One wave per 16x16 output tile; compile-time shapes so the K-loop is
// straight-line: 4x global_load_b128 for A + 16 scalar W loads per WMMA.
// Lane mapping (ISA 7.12.2, 16-bit A 16x32 / B 32x16):
//   vector element q   in [0,8)  <->  K = kk      + half*8 + q
//   vector element 8+q in [0,8)  <->  K = kk + 16 + half*8 + q

template <int CIN, int COUT>
__global__ void k_gemm_wmma_bf16(const float* __restrict__ X,   // nrows x CIN
                                 const float* __restrict__ Wm,  // CIN x COUT
                                 float* __restrict__ H,         // nrows x COUT
                                 int nrows) {
  constexpr int TILESN = COUT / 16;
  const int lane = threadIdx.x & 31;
  const int wave = threadIdx.x >> 5;
  const int tile = blockIdx.x * (blockDim.x >> 5) + wave;
  const int tm = tile / TILESN;
  const int tn = tile - tm * TILESN;
  if (tm * 16 >= nrows) return;        // wave-uniform: EXEC all-1s at WMMA

  const int half = lane >> 4;
  const int l16  = lane & 15;

  const float* __restrict__ Arow = X + (long long)(tm * 16 + l16) * CIN;
  const int bcol = tn * 16 + l16;

  v8f acc = {};
#pragma unroll
  for (int kk = 0; kk < CIN; kk += 32) {
    v16bf a, b;

    // ---- A: lane holds one row; two runs of 8 contiguous K values ----
    {
      const float4 x0 = *(const float4*)(Arow + kk + half * 8);
      const float4 x1 = *(const float4*)(Arow + kk + half * 8 + 4);
      a[0] = (__bf16)x0.x; a[1] = (__bf16)x0.y; a[2] = (__bf16)x0.z; a[3] = (__bf16)x0.w;
      a[4] = (__bf16)x1.x; a[5] = (__bf16)x1.y; a[6] = (__bf16)x1.z; a[7] = (__bf16)x1.w;
      if constexpr (CIN >= 32) {
        const float4 x2 = *(const float4*)(Arow + kk + 16 + half * 8);
        const float4 x3 = *(const float4*)(Arow + kk + 16 + half * 8 + 4);
        a[8]  = (__bf16)x2.x; a[9]  = (__bf16)x2.y; a[10] = (__bf16)x2.z; a[11] = (__bf16)x2.w;
        a[12] = (__bf16)x3.x; a[13] = (__bf16)x3.y; a[14] = (__bf16)x3.z; a[15] = (__bf16)x3.w;
      } else {
#pragma unroll
        for (int q = 0; q < 8; ++q) a[8 + q] = (__bf16)0.0f;
      }
    }

    // ---- B: lane holds one column; K strided by COUT (weights cache-resident) ----
#pragma unroll
    for (int q = 0; q < 8; ++q) {
      b[q] = (__bf16)Wm[(kk + half * 8 + q) * COUT + bcol];
      if constexpr (CIN >= 32)
        b[8 + q] = (__bf16)Wm[(kk + 16 + half * 8 + q) * COUT + bcol];
      else
        b[8 + q] = (__bf16)0.0f;
    }

    acc = __builtin_amdgcn_wmma_f32_16x16x32_bf16(
        /*neg_a=*/false, a, /*neg_b=*/false, b,
        /*c_mod=*/(short)0, acc, /*reuse_a=*/false, /*reuse_b=*/false);
  }

  // C/D layout: lanes 0-15 -> M = r, lanes 16-31 -> M = 8 + r; N = lane%16.
  const int orow0 = tm * 16 + half * 8;
#pragma unroll
  for (int r = 0; r < 8; ++r)
    H[(long long)(orow0 + r) * COUT + tn * 16 + l16] = acc[r];
}

// ---------------- SpMM scatter: AGG[dst] += val * H[src], 4 channels/thread ----------------

__global__ void k_spmm_scatter4(const int* __restrict__ src, const int* __restrict__ dst,
                                const float* __restrict__ vals, const float* __restrict__ H,
                                float* __restrict__ AGG, int E, int lc /* log2(C) */) {
  const int lcq = lc - 2;                       // log2(C/4) quads per edge
  long long t = (long long)blockIdx.x * blockDim.x + threadIdx.x;
  const long long total = (long long)E << lcq;
  if (t >= total) return;
  const int q = (int)(t & ((1 << lcq) - 1));    // quad index within row
  const int e = (int)(t >> lcq);
  const int s = src[e];
  const int d = dst[e];
  const float v = vals[e];
  const float4 h4 = *(const float4*)(H + ((long long)s << lc) + q * 4);
  float* ag = AGG + ((long long)d << lc) + q * 4;
  atomicAdd(ag + 0, v * h4.x);
  atomicAdd(ag + 1, v * h4.y);
  atomicAdd(ag + 2, v * h4.z);
  atomicAdd(ag + 3, v * h4.w);
}

// ---------------- epilogue: OUT = relu(AGG + self*H + b), float4 ----------------

__global__ void k_epilogue4(const float4* __restrict__ AGG, const float4* __restrict__ H,
                            const float* __restrict__ selfv, const float* __restrict__ bias,
                            float4* __restrict__ OUT, int N, int lc) {
  const int lcq = lc - 2;
  long long i = (long long)blockIdx.x * blockDim.x + threadIdx.x;
  const long long total = (long long)N << lcq;
  if (i >= total) return;
  const int q = (int)(i & ((1 << lcq) - 1));
  const int v = (int)(i >> lcq);
  const float sv = selfv[v];
  const float4 ag = AGG[i];
  const float4 h4 = H[i];
  const float4 b4 = *(const float4*)(bias + q * 4);
  float4 o;
  o.x = fmaxf(ag.x + sv * h4.x + b4.x, 0.0f);
  o.y = fmaxf(ag.y + sv * h4.y + b4.y, 0.0f);
  o.z = fmaxf(ag.z + sv * h4.z + b4.z, 0.0f);
  o.w = fmaxf(ag.w + sv * h4.w + b4.w, 0.0f);
  OUT[i] = o;
}

// ---------------------------------------------------------------------------

extern "C" void kernel_launch(void* const* d_in, const int* in_sizes, int n_in,
                              void* d_out, int out_size, void* d_ws, size_t ws_size,
                              hipStream_t stream) {
  const float* X0 = (const float*)d_in[0];          // N x 128
  const int*   EI = (const int*)d_in[1];            // 2 x E (int32)
  const float* W1 = (const float*)d_in[2]; const float* B1 = (const float*)d_in[3];
  const float* W2 = (const float*)d_in[4]; const float* B2 = (const float*)d_in[5];
  const float* W3 = (const float*)d_in[6]; const float* B3 = (const float*)d_in[7];
  const float* W4 = (const float*)d_in[8]; const float* B4 = (const float*)d_in[9];

  const int N = in_sizes[0] / 128;
  const int E = in_sizes[1] / 2;
  const int* esrc = EI;          // edge_index[0]
  const int* edst = EI + E;      // edge_index[1]

  float* out = (float*)d_out;
  float* x_recon = out;                       // N x 128
  float* z_out   = out + (long long)N * 128;  // N x 16

  // ---- workspace carve-up (floats) ----
  float* w = (float*)d_ws;
  float* deg   = w;              w += N;
  float* dis   = w;              w += N;
  float* dls   = w;              w += N;
  float* aself = w;              w += N;
  float* lself = w;              w += N;
  float* adjv  = w;              w += E;
  float* lapv  = w;              w += E;
  float* Hbuf  = w;              w += (long long)N * 128;
  float* Agg   = w;              w += (long long)N * 128;
  float* bufA  = w;              w += (long long)N * 32;   // 32-ch layer outputs

  const int TB = 256;
  auto blocks = [](long long n, int tb) { return (unsigned)((n + tb - 1) / tb); };

  // ---- preamble: degrees + normalization values ----
  k_fill4_f32<<<blocks(N / 4 + 1, TB), TB, 0, stream>>>((float4*)deg, 1.0f, (N + 3) / 4);
  k_count_deg<<<blocks(E, TB), TB, 0, stream>>>(edst, deg, E);
  k_norms<<<blocks(N, TB), TB, 0, stream>>>(deg, dis, dls, aself, lself, N);
  k_edge_vals<<<blocks(E, TB), TB, 0, stream>>>(esrc, edst, dis, dls, adjv, lapv, E);

  // ---- one GCN layer (GEMM dispatch is compile-time-shaped) ----
  auto layer_tail = [&](const float* bias, const float* evals, const float* sv,
                        int Cout, float* xout) {
    const int lc = __builtin_ctz(Cout);
    const long long agg4 = ((long long)N << lc) >> 2;
    k_fill4_f32<<<blocks(agg4, TB), TB, 0, stream>>>((float4*)Agg, 0.0f, agg4);
    const long long work = ((long long)E << lc) >> 2;
    k_spmm_scatter4<<<blocks(work, TB), TB, 0, stream>>>(esrc, edst, evals, Hbuf, Agg, E, lc);
    k_epilogue4<<<blocks(agg4, TB), TB, 0, stream>>>((const float4*)Agg, (const float4*)Hbuf,
                                                     sv, bias, (float4*)xout, N, lc);
  };
  auto gemm_grid = [&](int Cout) {
    const long long tiles = (long long)((N + 15) / 16) * (Cout / 16);
    return (unsigned)((tiles + 7) / 8);                                  // 8 waves / block
  };

  // x1 = relu(gcn(X0, W1; adj))          N x 32
  k_gemm_wmma_bf16<128, 32><<<gemm_grid(32), 256, 0, stream>>>(X0, W1, Hbuf, N);
  layer_tail(B1, adjv, aself, 32, bufA);

  // z  = relu(gcn(x1, W2; adj))          N x 16  -> d_out tail
  k_gemm_wmma_bf16<32, 16><<<gemm_grid(16), 256, 0, stream>>>(bufA, W2, Hbuf, N);
  layer_tail(B2, adjv, aself, 16, z_out);

  // x3 = relu(gcn(z,  W3; lap))          N x 32
  k_gemm_wmma_bf16<16, 32><<<gemm_grid(32), 256, 0, stream>>>(z_out, W3, Hbuf, N);
  layer_tail(B3, lapv, lself, 32, bufA);

  // x_recon = relu(gcn(x3, W4; lap))     N x 128 -> d_out head
  k_gemm_wmma_bf16<32, 128><<<gemm_grid(128), 256, 0, stream>>>(bufA, W4, Hbuf, N);
  layer_tail(B4, lapv, lself, 128, x_recon);
}
